// CausalMultiHeadAttention_86552180949270
// MI455X (gfx1250) — matrix operand order repaired and checked by reading, not compiled, not run on hardware
//
#include <hip/hip_runtime.h>

typedef __bf16 bf16;
typedef __attribute__((ext_vector_type(16))) __bf16 v16bf;
typedef __attribute__((ext_vector_type(8)))  __bf16 v8bf;
typedef __attribute__((ext_vector_type(4)))  __bf16 v4bf;
typedef __attribute__((ext_vector_type(2)))  __bf16 v2bf;
typedef __attribute__((ext_vector_type(8)))  float  v8f;
typedef __attribute__((ext_vector_type(4)))  float  v4f;

#define DMODEL 4096
#define NH     32
#define DK     128
#define SEQLEN 128
#define NEGBIG (-3.0e38f)

// ---------- helpers ----------

__device__ __forceinline__ bf16 f2bf(float f) {
  return (bf16)f;  // lowers to v_cvt_pk_bf16_f32 pairs on gfx1250 (RNE)
}

__device__ __forceinline__ v2bf pk2(float a, float b) {
  v2bf r; r[0] = f2bf(a); r[1] = f2bf(b); return r;
}

__device__ __forceinline__ v8f zero8() {
  v8f z;
#pragma unroll
  for (int i = 0; i < 8; ++i) z[i] = 0.0f;
  return z;
}

// A-fragment 16x32 bf16 from row-major source (ld in elements).
// lane<16: M=lane, K = k0..k0+15
// lane>=16: M=lane-16, K = k0+16..k0+31
__device__ __forceinline__ v16bf load_a_bf(const bf16* base, int ld, int row0,
                                           int k0, int lane) {
  int hf = lane >> 4, m = lane & 15;
  const bf16* p = base + (row0 + m) * ld;
  union { v16bf v; v8bf h[2]; } u;
  u.h[0] = *(const v8bf*)(p + k0 + hf * 16);
  u.h[1] = *(const v8bf*)(p + k0 + 8 + hf * 16);
  return u.v;
}

// B-fragment 32x16 bf16 for C = A * S^T, S row-major along K (ld elements).
// lane<16: N=lane, K=k0..k0+15 ; lane>=16: N=lane-16, K=k0+16..k0+31
__device__ __forceinline__ v16bf load_b_bf(const bf16* base, int ld, int col0,
                                           int k0, int lane) {
  int hf = lane >> 4, n = lane & 15;
  const bf16* p = base + (size_t)(col0 + n) * ld + k0 + hf * 16;
  return *(const v16bf*)p;
}

// Convert 16 raw fp32 (4 x v4f) into one bf16 fragment register block.
__device__ __forceinline__ v16bf cvt16(const v4f* q) {
  v16bf r;
#pragma unroll
  for (int i = 0; i < 4; ++i) {
#pragma unroll
    for (int j = 0; j < 4; ++j) r[i * 4 + j] = f2bf(q[i][j]);
  }
  return r;
}

__device__ __forceinline__ v8f wmma_bf(v16bf a, v16bf b, v8f c) {
  return __builtin_amdgcn_wmma_f32_16x16x32_bf16(
      false, a, false, b, (short)0, c, false, false);
}

// ---------- kernel 1: x fp32 -> bf16 ----------
__global__ void k_convert_x(const float* __restrict__ x, bf16* __restrict__ xb) {
  int idx = blockIdx.x * blockDim.x + threadIdx.x;  // 131072 threads * 4 elems
  int e = idx * 4;
  v4f f = *(const v4f*)(x + e);
  v4bf b;
#pragma unroll
  for (int i = 0; i < 4; ++i) b[i] = f2bf(f[i]);
  *(v4bf*)(xb + e) = b;
}

// ---------- kernel 2: QKV projection ----------
// 192 WGs x 128 threads: wg -> (p, h, half-of-d). Wave w owns a 16-col N strip
// over all 128 seq rows, so each W row is streamed from HBM exactly once.
// A-frags loaded as one clause per k-step; raw fp32 B double-buffered in regs.
__global__ void __launch_bounds__(128, 2)
k_qkv_proj(const float* __restrict__ Q, const float* __restrict__ K,
           const float* __restrict__ V, const bf16* __restrict__ xb,
           bf16* __restrict__ qp, bf16* __restrict__ kp,
           bf16* __restrict__ vpT) {
  int wg = blockIdx.x;                 // 0..191
  int p = wg >> 6;                     // 0..2
  int h = (wg >> 1) & 31;
  int nhalf = wg & 1;
  const float* W = (p == 0 ? Q : (p == 1 ? K : V)) + (size_t)h * DK * DMODEL;
  int wave = threadIdx.x >> 5, lane = threadIdx.x & 31;
  int hf = lane >> 4, nlow = lane & 15;
  int n0 = nhalf * 64 + wave * 16;
  const float* Wrow = W + (size_t)(n0 + nlow) * DMODEL + hf * 16;  // B source

  v8f acc[8];
#pragma unroll
  for (int i = 0; i < 8; ++i) acc[i] = zero8();

  v16bf aF[8];
  v4f bCur[4], bNxt[4];
#pragma unroll
  for (int i = 0; i < 4; ++i) bCur[i] = *(const v4f*)(Wrow + 4 * i);

  for (int k0 = 0; k0 < DMODEL - 32; k0 += 32) {
    // current step's A fragments: 16 x b128 in one clause, one wait total
#pragma unroll
    for (int mt = 0; mt < 8; ++mt)
      aF[mt] = load_a_bf(xb, DMODEL, mt * 16, k0, lane);
    // next step's raw B row
#pragma unroll
    for (int i = 0; i < 4; ++i) bNxt[i] = *(const v4f*)(Wrow + k0 + 32 + 4 * i);
    __builtin_prefetch(Wrow + k0 + 64, 0, 3);
    v16bf b = cvt16(bCur);
#pragma unroll
    for (int mt = 0; mt < 8; ++mt) acc[mt] = wmma_bf(aF[mt], b, acc[mt]);
#pragma unroll
    for (int i = 0; i < 4; ++i) bCur[i] = bNxt[i];
  }
  {  // epilogue k0 = DMODEL-32
#pragma unroll
    for (int mt = 0; mt < 8; ++mt)
      aF[mt] = load_a_bf(xb, DMODEL, mt * 16, DMODEL - 32, lane);
    v16bf b = cvt16(bCur);
#pragma unroll
    for (int mt = 0; mt < 8; ++mt) acc[mt] = wmma_bf(aF[mt], b, acc[mt]);
  }

  // C layout: VGPR r -> M = r + 8*(lane>>4), N = lane&15
  int mloc = hf * 8;
  int d = n0 + nlow;
  if (p == 2) {
    // vpT[h][d][s]; r-consecutive s -> packed 16B stores
#pragma unroll
    for (int mt = 0; mt < 8; ++mt) {
      union { v8bf v; v2bf c[4]; } u;
#pragma unroll
      for (int j = 0; j < 4; ++j) u.c[j] = pk2(acc[mt][2 * j], acc[mt][2 * j + 1]);
      *(v8bf*)(vpT + (size_t)h * DK * SEQLEN + d * SEQLEN + mt * 16 + mloc) = u.v;
    }
  } else {
    bf16* dst = (p == 0) ? qp : kp;
#pragma unroll
    for (int mt = 0; mt < 8; ++mt) {
#pragma unroll
      for (int r = 0; r < 8; ++r) {
        int s = mt * 16 + mloc + r;
        dst[(size_t)h * SEQLEN * DK + s * DK + d] = f2bf(acc[mt][r]);
      }
    }
  }
}

// ---------- kernel 3: per-head attention ----------
// Row-strip partition: wave w owns rows q=16w..16w+15; causal mask + softmax
// stay in registers (shfl_xor over the 16-lane half); probs round-trip LDS.
__global__ void __launch_bounds__(256, 2)
k_attention(const bf16* __restrict__ qp, const bf16* __restrict__ kp,
            const bf16* __restrict__ vpT, bf16* __restrict__ attT) {
  __shared__ bf16 probs[SEQLEN][SEQLEN];  // 32 KiB
  int h = blockIdx.x;
  int wave = threadIdx.x >> 5, lane = threadIdx.x & 31;
  int hf = lane >> 4, col = lane & 15;
  const bf16* qh = qp + (size_t)h * SEQLEN * DK;
  const bf16* kh = kp + (size_t)h * SEQLEN * DK;
  const bf16* vh = vpT + (size_t)h * DK * SEQLEN;
  int m0 = wave * 16;

  v8f acc[8];
#pragma unroll
  for (int i = 0; i < 8; ++i) acc[i] = zero8();
  for (int k0 = 0; k0 < DK; k0 += 32) {
    v16bf a = load_a_bf(qh, DK, m0, k0, lane);
#pragma unroll
    for (int nt = 0; nt < 8; ++nt) {
      v16bf b = load_b_bf(kh, DK, nt * 16, k0, lane);
      acc[nt] = wmma_bf(a, b, acc[nt]);
    }
  }

  const float scale = 0.08838834764831845f;  // 1/sqrt(128)
#pragma unroll
  for (int r = 0; r < 8; ++r) {
    int q = m0 + r + 8 * hf;
    float v[8];
    float mx = NEGBIG;
#pragma unroll
    for (int nt = 0; nt < 8; ++nt) {
      int k = nt * 16 + col;
      float s = acc[nt][r] * scale;
      bool keep = (k >= q) && (s != 0.0f);  // triu mask, masked==0 -> -inf
      v[nt] = keep ? s : NEGBIG;
      mx = fmaxf(mx, v[nt]);
    }
#pragma unroll
    for (int sft = 1; sft < 16; sft <<= 1) mx = fmaxf(mx, __shfl_xor(mx, sft, 32));
    float sum = 0.0f;
#pragma unroll
    for (int nt = 0; nt < 8; ++nt) {
      v[nt] = __expf(v[nt] - mx);
      sum += v[nt];
    }
#pragma unroll
    for (int sft = 1; sft < 16; sft <<= 1) sum += __shfl_xor(sum, sft, 32);
    float inv = 1.0f / sum;
#pragma unroll
    for (int nt = 0; nt < 8; ++nt)
      probs[q][nt * 16 + col] = f2bf(v[nt] * inv);
  }
  __syncthreads();

  v8f acc2[8];
#pragma unroll
  for (int i = 0; i < 8; ++i) acc2[i] = zero8();
  for (int k0 = 0; k0 < SEQLEN; k0 += 32) {
    v16bf a = load_a_bf(&probs[0][0], SEQLEN, m0, k0, lane);
#pragma unroll
    for (int nt = 0; nt < 8; ++nt) {
      v16bf b = load_b_bf(vh, SEQLEN, nt * 16, k0, lane);
      acc2[nt] = wmma_bf(a, b, acc2[nt]);
    }
  }
  // att^T: attT[d][h*128 + q]; r-consecutive q -> packed 16B stores
#pragma unroll
  for (int nt = 0; nt < 8; ++nt) {
    int d = nt * 16 + col;
    union { v8bf v; v2bf c[4]; } u;
#pragma unroll
    for (int j = 0; j < 4; ++j) u.c[j] = pk2(acc2[nt][2 * j], acc2[nt][2 * j + 1]);
    *(v8bf*)(attT + (size_t)d * (NH * SEQLEN) + h * SEQLEN + m0 + 8 * hf) = u.v;
  }
}

// ---------- kernel 4a: partial out = O_kslice @ att_kslice ----------
// 128 WGs: (32 M-blocks) x (4 K-slices of 1024). O staged fp32->bf16 through
// double-buffered LDS (read once from HBM); partials to workspace (no float
// atomics -> deterministic).
__global__ void __launch_bounds__(256, 2)
k_out_partial(const float* __restrict__ O, const bf16* __restrict__ attT,
              float* __restrict__ part) {
  __shared__ bf16 Atile[2][128][32];  // 2 x 8 KiB
  int mblk = (blockIdx.x >> 2) * 128;
  int kslice = blockIdx.x & 3;
  int kbeg = kslice * 1024, kend = kbeg + 1024;
  int wave = threadIdx.x >> 5, lane = threadIdx.x & 31;
  int n0 = wave * 16;

  v8f acc[8];
#pragma unroll
  for (int i = 0; i < 8; ++i) acc[i] = zero8();

  auto stage = [&](int buf, int k0) {
#pragma unroll
    for (int it = 0; it < 4; ++it) {
      int e = threadIdx.x + it * 256;  // 1024 float4 chunks (128 rows x 32 cols)
      int row = e >> 3;
      int c4 = (e & 7) * 4;
      v4f f = *(const v4f*)(O + (size_t)(mblk + row) * DMODEL + k0 + c4);
      v4bf b4;
#pragma unroll
      for (int j = 0; j < 4; ++j) b4[j] = f2bf(f[j]);
      *(v4bf*)(&Atile[buf][row][c4]) = b4;
    }
  };

  stage(0, kbeg);
  __syncthreads();
  int cur = 0;
  for (int k0 = kbeg; k0 < kend - 32; k0 += 32) {
    stage(cur ^ 1, k0 + 32);  // overlap next-tile staging with current wmmas
    v16bf b = load_b_bf(attT, NH * SEQLEN, n0, k0, lane);
#pragma unroll
    for (int mt = 0; mt < 8; ++mt) {
      v16bf a = load_a_bf(&Atile[cur][0][0], 32, mt * 16, 0, lane);
      acc[mt] = wmma_bf(a, b, acc[mt]);
    }
    __syncthreads();
    cur ^= 1;
  }
  {  // epilogue: k0 = kend-32, buffer `cur` staged by last loop iteration
    v16bf b = load_b_bf(attT, NH * SEQLEN, n0, kend - 32, lane);
#pragma unroll
    for (int mt = 0; mt < 8; ++mt) {
      v16bf a = load_a_bf(&Atile[cur][0][0], 32, mt * 16, 0, lane);
      acc[mt] = wmma_bf(a, b, acc[mt]);
    }
  }

  int n = lane & 15, mloc = (lane >> 4) * 8;
  float* dst = part + (size_t)kslice * (DMODEL * DK);
#pragma unroll
  for (int mt = 0; mt < 8; ++mt) {
#pragma unroll
    for (int r = 0; r < 8; ++r) {
      int i = mblk + mt * 16 + mloc + r;
      dst[(size_t)i * DK + n0 + n] = acc[mt][r];
    }
  }
}

// ---------- kernel 4b: deterministic fixed-order reduction ----------
__global__ void k_out_reduce(const float* __restrict__ part, float* __restrict__ out) {
  int idx = blockIdx.x * blockDim.x + threadIdx.x;  // 131072 threads
  int e = idx * 4;
  const size_t S = (size_t)DMODEL * DK;
  v4f s = *(const v4f*)(part + e);
#pragma unroll
  for (int kslice = 1; kslice < 4; ++kslice) {
    v4f f = *(const v4f*)(part + kslice * S + e);
#pragma unroll
    for (int j = 0; j < 4; ++j) s[j] += f[j];
  }
  *(v4f*)(out + e) = s;
}

// ---------- launch ----------
extern "C" void kernel_launch(void* const* d_in, const int* in_sizes, int n_in,
                              void* d_out, int out_size, void* d_ws, size_t ws_size,
                              hipStream_t stream) {
  (void)in_sizes; (void)n_in; (void)out_size; (void)ws_size;
  const float* Q = (const float*)d_in[0];
  const float* K = (const float*)d_in[1];
  const float* V = (const float*)d_in[2];
  const float* O = (const float*)d_in[3];
  const float* x = (const float*)d_in[4];
  float* out = (float*)d_out;

  char* w = (char*)d_ws;
  const size_t MB = 1u << 20;
  bf16* xb    = (bf16*)(w + 0 * MB);  // 128x4096 bf16          (1 MiB)
  bf16* qp    = (bf16*)(w + 1 * MB);  // 32 x 128x128 (s,d)     (1 MiB)
  bf16* kp    = (bf16*)(w + 2 * MB);  // 32 x 128x128 (s,d)     (1 MiB)
  bf16* vpT   = (bf16*)(w + 3 * MB);  // 32 x 128x128 (d,s)     (1 MiB)
  bf16* attT  = (bf16*)(w + 4 * MB);  // 128 x 4096 (d, h*s)    (1 MiB)
  float* part = (float*)(w + 5 * MB); // 4 x 4096x128 f32       (8 MiB)

  k_convert_x<<<512, 256, 0, stream>>>(x, xb);
  k_qkv_proj<<<192, 128, 0, stream>>>(Q, K, V, xb, qp, kp, vpT);
  k_attention<<<32, 256, 0, stream>>>(qp, kp, vpT, attT);
  k_out_partial<<<128, 256, 0, stream>>>(O, attT, part);
  k_out_reduce<<<512, 256, 0, stream>>>(part, out);
}
